// RoIHeader_63797444215094
// MI455X (gfx1250) — matrix operand order repaired and checked by reading, not compiled, not run on hardware
//
#include <hip/hip_runtime.h>
#include <hip/hip_bf16.h>

typedef __attribute__((ext_vector_type(16))) __bf16 bf16x16;
typedef __attribute__((ext_vector_type(8)))  float  f32x8;

__device__ __forceinline__ unsigned short f2bf(float f) {
  unsigned int u = __float_as_uint(f);
  unsigned int r = u + 0x7FFFu + ((u >> 16) & 1u);   // round-to-nearest-even
  return (unsigned short)(r >> 16);
}

// ---------------------------------------------------------------------------
// Elementwise f32 -> bf16 (weights)
// ---------------------------------------------------------------------------
__global__ void f32_to_bf16_kernel(const float* __restrict__ in,
                                   unsigned short* __restrict__ out, int n) {
  int i = blockIdx.x * 256 + threadIdx.x;
  if (i < n) out[i] = f2bf(in[i]);
}

// ---------------------------------------------------------------------------
// RoIAlign: 1000 blocks x 256 threads (1 thread per channel).
// Writes bf16 A-matrix [1000, 12544], K index = c*49 + ph*7 + pw.
// ---------------------------------------------------------------------------
__global__ void roi_align_kernel(const float* __restrict__ p2,
                                 const float* __restrict__ p3,
                                 const float* __restrict__ p4,
                                 const float* __restrict__ p5,
                                 const float* __restrict__ rois,
                                 const int* __restrict__ img_sz,
                                 unsigned short* __restrict__ Abf) {
  const int n = blockIdx.x;
  const int c = threadIdx.x;
  float x1 = rois[n * 4 + 0], y1 = rois[n * 4 + 1];
  float x2 = rois[n * 4 + 2], y2 = rois[n * 4 + 3];
  float h = y2 - y1, w = x2 - x1;
  float img = (float)(*img_sz);
  // roi_level = 4 + log2(sqrt(h*w) / (224/img))
  float lvlf = 4.0f + __log2f(sqrtf(fmaxf(h * w, 1e-20f)) * img * (1.0f / 224.0f));
  int lvl = (int)rintf(lvlf);
  lvl = min(max(lvl, 2), 5) - 2;
  const int S = 256 >> lvl;
  const float* feat =
      (lvl == 0 ? p2 : lvl == 1 ? p3 : lvl == 2 ? p4 : p5) + (size_t)c * S * S;
  const float sc = (float)S;
  const float x1s = x1 * sc, y1s = y1 * sc;
  const float rw = fmaxf(x2 * sc - x1s, 1.0f);
  const float rh = fmaxf(y2 * sc - y1s, 1.0f);
  const float inv14 = 1.0f / 14.0f;

  for (int ph = 0; ph < 7; ++ph) {
    for (int pw = 0; pw < 7; ++pw) {
      float sum = 0.f;
#pragma unroll
      for (int sy = 0; sy < 2; ++sy) {
#pragma unroll
        for (int sx = 0; sx < 2; ++sx) {
          float yy = y1s + rh * ((float)(2 * ph + sy) + 0.5f) * inv14;
          float xx = x1s + rw * ((float)(2 * pw + sx) + 0.5f) * inv14;
          bool valid = (yy >= -1.f) && (yy <= sc) && (xx >= -1.f) && (xx <= sc);
          float y = fminf(fmaxf(yy, 0.f), sc - 1.f);
          float x = fminf(fmaxf(xx, 0.f), sc - 1.f);
          int y0 = (int)floorf(y), x0 = (int)floorf(x);
          int y1i = min(y0 + 1, S - 1), x1i = min(x0 + 1, S - 1);
          float ly = y - (float)y0, lx = x - (float)x0;
          float hy = 1.f - ly, hx = 1.f - lx;
          float v = hy * hx * feat[y0 * S + x0] + hy * lx * feat[y0 * S + x1i] +
                    ly * hx * feat[y1i * S + x0] + ly * lx * feat[y1i * S + x1i];
          sum += valid ? v : 0.f;
        }
      }
      Abf[(size_t)n * 12544 + c * 49 + ph * 7 + pw] = f2bf(sum * 0.25f);
    }
  }
}

// ---------------------------------------------------------------------------
// bf16 GEMM with WMMA: C[M,O] = A[M,K] * W[O,K]^T, fused epilogue.
// Block: 256 threads = 8 wave32s. Block tile 64(M) x 128(N), K step 32.
// Double-buffered LDS: tile k+1 staged (async A copy via ASYNCcnt + v_perm
// pair-packed B transpose) while tile k is consumed; one barrier/iteration.
// All 5 fragments loaded before the 4 back-to-back WMMAs.
// MODE 0: y = relu((x + b - mean)*rsqrt(var+eps)*gamma + beta) -> bf16 out
// MODE 1: y = x + bias, scattered to logits (col<81) / bbox (col>=81), f32
// ---------------------------------------------------------------------------
#define BM 64
#define BN 128
#define BK 32
#define AK 40    // padded LDS row stride for A tile (ushorts): bank-spread
#define BNP 136  // padded LDS row stride for B tile (ushorts): 272B, 16B-aligned

union FragU {
  uint4 u[2];
  bf16x16 v;
};

template <int MODE>
__global__ __launch_bounds__(256) void gemm_bf16_kernel(
    const unsigned short* __restrict__ A, const unsigned short* __restrict__ W,
    int M, int K, int O,
    const float* __restrict__ bias, const float* __restrict__ gamma,
    const float* __restrict__ beta, const float* __restrict__ mean,
    const float* __restrict__ var, unsigned short* __restrict__ outBf,
    const float* __restrict__ bcls, const float* __restrict__ bbbox,
    float* __restrict__ outLogits, float* __restrict__ outBbox) {
  __shared__ unsigned short As[2][BM * AK];
  __shared__ unsigned short Bs[2][BK * BNP];

  const int t = threadIdx.x;
  const int lane = t & 31, wid = t >> 5;
  const int hf = lane >> 4, mr = lane & 15;
  const int waveM = wid >> 1, waveN = wid & 1;
  const int mTile = blockIdx.x * BM;
  const int oTile = blockIdx.y * BN;

  f32x8 acc[4];
#pragma unroll
  for (int i = 0; i < 4; ++i)
#pragma unroll
    for (int j = 0; j < 8; ++j) acc[i][j] = 0.f;

  // ---- A staging task: one 16B chunk per thread (clamped row, no guard) ----
  const int am = t >> 2;        // 0..63 : A tile row
  const int akc = (t & 3) * 8;  // K chunk within step
  const int aRow = min(mTile + am, M - 1);
  const unsigned short* aSrc = A + (size_t)aRow * (size_t)K + akc;
  const unsigned int aLds0 = (unsigned int)(size_t)(&As[0][am * AK + akc]);
  const unsigned int aLds1 = (unsigned int)(size_t)(&As[1][am * AK + akc]);

  // ---- B staging task: two adjacent weight rows, 8 k's, pair-packed ----
  const int bp2 = t >> 2;             // 0..63 : column pair index
  const int bkc = (t & 3) * 8;        // K chunk
  const int o0 = min(oTile + 2 * bp2, O - 1);
  const int o1 = min(oTile + 2 * bp2 + 1, O - 1);
  const unsigned short* w0Src = W + (size_t)o0 * (size_t)K + bkc;
  const unsigned short* w1Src = W + (size_t)o1 * (size_t)K + bkc;

  auto stage = [&](int kk, int buf) {
    // A tile: async global -> LDS copy (16B per lane), tracked by ASYNCcnt
    asm volatile("global_load_async_to_lds_b128 %0, %1, off"
                 :: "v"(buf ? aLds1 : aLds0), "v"(aSrc + kk) : "memory");
    // B tile: 2 rows x 8 k's; v_perm interleave {w0[k], w1[k]} -> ds_store_b32
    uint4 w0 = *(const uint4*)(w0Src + kk);
    uint4 w1 = *(const uint4*)(w1Src + kk);
    const unsigned int* a32 = (const unsigned int*)&w0;
    const unsigned int* b32 = (const unsigned int*)&w1;
    unsigned short* BsBuf = &Bs[buf][0];
#pragma unroll
    for (int j2 = 0; j2 < 4; ++j2) {
      unsigned int lo = __builtin_amdgcn_perm(b32[j2], a32[j2], 0x05040100u);
      unsigned int hi = __builtin_amdgcn_perm(b32[j2], a32[j2], 0x07060302u);
      *(unsigned int*)(&BsBuf[(bkc + 2 * j2) * BNP + 2 * bp2]) = lo;
      *(unsigned int*)(&BsBuf[(bkc + 2 * j2 + 1) * BNP + 2 * bp2]) = hi;
    }
  };

  // prologue: stage first tile into buffer 0
  stage(0, 0);
  asm volatile("s_wait_asynccnt 0x0" ::: "memory");
  __syncthreads();

  for (int k0 = 0; k0 < K; k0 += BK) {
    const int cur = (k0 / BK) & 1;
    const int nxt = cur ^ 1;

    // stage next tile while this one is consumed
    if (k0 + BK < K) {
      stage(k0 + BK, nxt);
      if (k0 + 2 * BK < K) {  // prefetch the tile after next (global_prefetch_b8)
        __builtin_prefetch((const void*)(w0Src + k0 + 2 * BK), 0, 1);
        __builtin_prefetch((const void*)(w1Src + k0 + 2 * BK), 0, 1);
      }
    }

    // ---- load all fragments for the current tile ----
    const unsigned short* ap = &As[cur][(waveM * 16 + mr) * AK];
    FragU fa;
    fa.u[0] = *(const uint4*)(ap + hf * 8);
    fa.u[1] = *(const uint4*)(ap + 16 + hf * 8);

    const unsigned short* bp = &Bs[cur][waveN * 64 + mr];
    FragU fb[4];
#pragma unroll
    for (int nsub = 0; nsub < 4; ++nsub) {
#pragma unroll
      for (int s = 0; s < 16; ++s) {
        const int kb = (s < 8) ? (hf * 8 + s) : (16 + hf * 8 + (s - 8));
        fb[nsub].v[s] = __builtin_bit_cast(__bf16, bp[kb * BNP + nsub * 16]);
      }
    }

    // ---- 4 back-to-back WMMAs ----
#pragma unroll
    for (int nsub = 0; nsub < 4; ++nsub) {
      acc[nsub] = __builtin_amdgcn_wmma_f32_16x16x32_bf16(
          false, fa.v, false, fb[nsub].v, (short)0, acc[nsub], false, false);
    }

    // own async copy complete, then publish staged tile / release current
    asm volatile("s_wait_asynccnt 0x0" ::: "memory");
    __syncthreads();
  }

  // ---- epilogue ----
  const int rowBase = mTile + waveM * 16 + hf * 8;
#pragma unroll
  for (int nsub = 0; nsub < 4; ++nsub) {
    int col = oTile + waveN * 64 + nsub * 16 + mr;
    if (col >= O) continue;
    if (MODE == 0) {
      float sc = rsqrtf(var[col] + 1e-3f) * gamma[col];
      float add = (bias[col] - mean[col]) * sc + beta[col];
#pragma unroll
      for (int i = 0; i < 8; ++i) {
        int row = rowBase + i;
        if (row < M) {
          float y = acc[nsub][i] * sc + add;
          y = fmaxf(y, 0.f);
          outBf[(size_t)row * O + col] = f2bf(y);
        }
      }
    } else {
      float b = (col < 81) ? bcls[col] : bbbox[col - 81];
#pragma unroll
      for (int i = 0; i < 8; ++i) {
        int row = rowBase + i;
        if (row < M) {
          float y = acc[nsub][i] + b;
          if (col < 81)
            outLogits[(size_t)row * 81 + col] = y;
          else
            outBbox[(size_t)row * 324 + (col - 81)] = y;
        }
      }
    }
  }
}

// ---------------------------------------------------------------------------
// Softmax over 81 classes, one wave32 per row.
// ---------------------------------------------------------------------------
__global__ void softmax81_kernel(const float* __restrict__ logits,
                                 float* __restrict__ probs) {
  const int r = blockIdx.x;
  const int lane = threadIdx.x;
  float v[3];
  float mx = -3.4e38f;
#pragma unroll
  for (int j = 0; j < 3; ++j) {
    int c = lane + 32 * j;
    v[j] = (c < 81) ? logits[(size_t)r * 81 + c] : -3.4e38f;
    mx = fmaxf(mx, v[j]);
  }
#pragma unroll
  for (int off = 16; off > 0; off >>= 1) mx = fmaxf(mx, __shfl_xor(mx, off, 32));
  float e[3];
  float s = 0.f;
#pragma unroll
  for (int j = 0; j < 3; ++j) {
    int c = lane + 32 * j;
    e[j] = (c < 81) ? __expf(v[j] - mx) : 0.f;
    s += e[j];
  }
#pragma unroll
  for (int off = 16; off > 0; off >>= 1) s += __shfl_xor(s, off, 32);
  float inv = 1.0f / s;
#pragma unroll
  for (int j = 0; j < 3; ++j) {
    int c = lane + 32 * j;
    if (c < 81) probs[(size_t)r * 81 + c] = e[j] * inv;
  }
}

// ---------------------------------------------------------------------------
extern "C" void kernel_launch(void* const* d_in, const int* in_sizes, int n_in,
                              void* d_out, int out_size, void* d_ws,
                              size_t ws_size, hipStream_t stream) {
  const float* p2 = (const float*)d_in[0];
  const float* p3 = (const float*)d_in[1];
  const float* p4 = (const float*)d_in[2];
  const float* p5 = (const float*)d_in[3];
  const float* rois = (const float*)d_in[4];
  const float* w1 = (const float*)d_in[5];
  const float* b1 = (const float*)d_in[6];
  const float* g1 = (const float*)d_in[7];
  const float* be1 = (const float*)d_in[8];
  const float* m1 = (const float*)d_in[9];
  const float* v1 = (const float*)d_in[10];
  const float* w2 = (const float*)d_in[11];
  const float* b2 = (const float*)d_in[12];
  const float* g2 = (const float*)d_in[13];
  const float* be2 = (const float*)d_in[14];
  const float* m2 = (const float*)d_in[15];
  const float* v2 = (const float*)d_in[16];
  const float* wcls = (const float*)d_in[17];
  const float* bcls = (const float*)d_in[18];
  const float* wbb = (const float*)d_in[19];
  const float* bbb = (const float*)d_in[20];
  const int* isz = (const int*)d_in[21];

  const int N = 1000, K1 = 12544, HID = 1024, OC = 405;

  char* ws = (char*)d_ws;
  size_t off = 0;
  auto alloc = [&](size_t bytes) -> void* {
    void* p = ws + off;
    off += (bytes + 255) & ~(size_t)255;
    return p;
  };
  unsigned short* A1  = (unsigned short*)alloc((size_t)1024 * K1 * 2);
  unsigned short* W1b = (unsigned short*)alloc((size_t)1024 * K1 * 2);
  unsigned short* W2b = (unsigned short*)alloc((size_t)1024 * 1024 * 2);
  unsigned short* W3b = (unsigned short*)alloc((size_t)512 * 1024 * 2);
  unsigned short* H1  = (unsigned short*)alloc((size_t)1024 * 1024 * 2);
  unsigned short* H2  = (unsigned short*)alloc((size_t)1024 * 1024 * 2);

  float* outLogits = (float*)d_out;
  float* outProbs  = (float*)d_out + 81000;
  float* outBbox   = (float*)d_out + 162000;

  // weight casts to bf16
  int n1 = 1024 * K1;
  f32_to_bf16_kernel<<<(n1 + 255) / 256, 256, 0, stream>>>(w1, W1b, n1);
  int n2 = 1024 * 1024;
  f32_to_bf16_kernel<<<(n2 + 255) / 256, 256, 0, stream>>>(w2, W2b, n2);
  int n3 = 81 * 1024;
  f32_to_bf16_kernel<<<(n3 + 255) / 256, 256, 0, stream>>>(wcls, W3b, n3);
  int n4 = 324 * 1024;
  f32_to_bf16_kernel<<<(n4 + 255) / 256, 256, 0, stream>>>(
      wbb, W3b + (size_t)81 * 1024, n4);

  // RoIAlign -> bf16 A matrix
  roi_align_kernel<<<N, 256, 0, stream>>>(p2, p3, p4, p5, rois, isz, A1);

  // FC1 + BN + ReLU
  dim3 gA((N + BM - 1) / BM, HID / BN);  // 16 x 8
  gemm_bf16_kernel<0><<<gA, 256, 0, stream>>>(A1, W1b, N, K1, HID, b1, g1, be1,
                                              m1, v1, H1, nullptr, nullptr,
                                              nullptr, nullptr);
  // FC2 + BN + ReLU
  gemm_bf16_kernel<0><<<gA, 256, 0, stream>>>(H1, W2b, N, HID, HID, b2, g2,
                                              be2, m2, v2, H2, nullptr, nullptr,
                                              nullptr, nullptr);
  // cls + bbox heads (fused, O=405)
  dim3 gB((N + BM - 1) / BM, (OC + BN - 1) / BN);  // 16 x 4
  gemm_bf16_kernel<1><<<gB, 256, 0, stream>>>(H2, W3b, N, HID, OC, nullptr,
                                              nullptr, nullptr, nullptr,
                                              nullptr, nullptr, bcls, bbb,
                                              outLogits, outBbox);
  // softmax
  softmax81_kernel<<<N, 32, 0, stream>>>(outLogits, outProbs);
}
